// Vim_CNN_35467839930876
// MI455X (gfx1250) — compile-verified, matrix operand order checked by reading, not run on hardware
//
#include <hip/hip_runtime.h>
#include <hip/hip_bf16.h>

// ---------------------------------------------------------------------------
// Vim block (CNN + Mamba mixer) for MI455X / gfx1250.
// All GEMMs run on v_wmma_f32_16x16x32_bf16 (wave32, 64x64 workgroup tiles,
// double-buffered LDS staged via GLOBAL_LOAD_ASYNC_TO_LDS_B128 when the
// toolchain exposes it). Scan + pointwise in f32 VALU.
// ---------------------------------------------------------------------------

typedef __bf16 bf16_t;
typedef __attribute__((ext_vector_type(16))) __bf16 v16bf;
typedef __attribute__((ext_vector_type(8)))  __bf16 v8bf;
typedef __attribute__((ext_vector_type(8)))  float  v8f;
typedef int v4i __attribute__((vector_size(16)));   // matches async-LDS builtin

#define BB 4
#define CC 256
#define HH 48
#define WW 48
#define LL (HH * WW)          // 2304
#define MM (BB * LL)          // 9216
#define DIN 512               // D_INNER
#define DST 16                // D_STATE
#define DTR 16                // DT_RANK

#define AS1 __attribute__((address_space(1)))
#define AS3 __attribute__((address_space(3)))

#if defined(__HIP_DEVICE_COMPILE__) && \
    __has_builtin(__builtin_amdgcn_global_load_async_to_lds_b128) && \
    __has_builtin(__builtin_amdgcn_s_wait_asynccnt)
#define HAVE_ASYNC_LDS 1
#else
#define HAVE_ASYNC_LDS 0
#endif

// ---------------- helpers ----------------

__device__ __forceinline__ bf16_t f2bf(float f) {
  unsigned int u = __builtin_bit_cast(unsigned int, f);
  u += 0x7fffu + ((u >> 16) & 1u);           // round-to-nearest-even
  unsigned short h = (unsigned short)(u >> 16);
  return __builtin_bit_cast(bf16_t, h);
}

__device__ __forceinline__ void wait_async_lds() {
#if HAVE_ASYNC_LDS
  __builtin_amdgcn_s_wait_asynccnt(0);
#endif
}

// Load one 16x32 bf16 fragment (rows x K) from an LDS tile with row stride
// `ld` elements, matching the CDNA5 WMMA A/B VGPR layout:
//   lane<16 : row = lane,    K = {0..7, 16..23}
//   lane>=16: row = lane-16, K = {8..15, 24..31}
__device__ __forceinline__ v16bf load_frag(const bf16_t* p, int ld, int lane) {
  int r  = lane & 15;
  int kb = (lane >> 4) << 3;                 // 0 or 8
  const bf16_t* q = p + r * ld + kb;
  v8bf lo = *(const v8bf*)(q);
  v8bf hi = *(const v8bf*)(q + 16);
  return __builtin_shufflevector(lo, hi, 0, 1, 2, 3, 4, 5, 6, 7,
                                 8, 9, 10, 11, 12, 13, 14, 15);
}

// ---------------- weight precast kernels ----------------

__global__ void cast_nk_kernel(const float* __restrict__ src,
                               bf16_t* __restrict__ dst, int count) {
  int i = blockIdx.x * blockDim.x + threadIdx.x;
  if (i < count) dst[i] = f2bf(src[i]);
}

// conv 3x3 weight (N=256, K=256, 3, 3) -> (N, K'=2304), K' ordered (tap, c)
__global__ void cast_conv3_kernel(const float* __restrict__ src,
                                  bf16_t* __restrict__ dst) {
  int i = blockIdx.x * blockDim.x + threadIdx.x;
  if (i >= CC * 9 * CC) return;
  int n = i / (9 * CC);
  int r = i % (9 * CC);
  int t = r / CC;
  int c = r % CC;
  dst[i] = f2bf(src[(n * CC + c) * 9 + t]);
}

// dt_proj weight (512,16) -> (512,32), K padded with zeros
__global__ void cast_dtw_kernel(const float* __restrict__ src,
                                bf16_t* __restrict__ dst) {
  int i = blockIdx.x * blockDim.x + threadIdx.x;
  if (i >= DIN * 32) return;
  int n = i / 32, k = i % 32;
  dst[i] = (k < DTR) ? f2bf(src[n * DTR + k]) : f2bf(0.0f);
}

// ---------------- layernorm + layout change ----------------
// One wave per spatial row m=(b,h,w): reads x[b,c,h,w] (stride L over c),
// writes xb[m,c] (raw bf16, for the conv path) and hs[m,c] (layernormed bf16).
__global__ __launch_bounds__(256) void ln_cast_kernel(
    const float* __restrict__ x, const float* __restrict__ g,
    const float* __restrict__ bta, bf16_t* __restrict__ xb,
    bf16_t* __restrict__ hs) {
  int wid = threadIdx.x >> 5, lane = threadIdx.x & 31;
  int m = blockIdx.x * 8 + wid;
  int b = m / LL, hw = m % LL;
  const float* xp = x + (size_t)b * CC * LL + hw;
  float v[8];
  float s = 0.f, s2 = 0.f;
#pragma unroll
  for (int j = 0; j < 8; ++j) {
    int c = lane + j * 32;
    float t = xp[(size_t)c * LL];
    v[j] = t; s += t; s2 += t * t;
  }
#pragma unroll
  for (int o = 16; o > 0; o >>= 1) {
    s += __shfl_xor(s, o, 32);
    s2 += __shfl_xor(s2, o, 32);
  }
  float mu = s * (1.0f / CC);
  float var = s2 * (1.0f / CC) - mu * mu;
  float rs = rsqrtf(var + 1e-5f);
#pragma unroll
  for (int j = 0; j < 8; ++j) {
    int c = lane + j * 32;
    xb[(size_t)m * CC + c] = f2bf(v[j]);
    hs[(size_t)m * CC + c] = f2bf((v[j] - mu) * rs * g[c] + bta[c]);
  }
}

// ---------------- im2col for 3x3, pad 1 ----------------
// src: (M, 256) in (b,h,w,c);  dst: (M, 2304), col = tap*256 + c
__global__ void im2col3x3_kernel(const bf16_t* __restrict__ src,
                                 bf16_t* __restrict__ dst) {
  int i = blockIdx.x * blockDim.x + threadIdx.x;     // over M * 288 vec8
  if (i >= MM * 288) return;
  int m = i / 288;
  int col = (i % 288) * 8;
  int t = col / CC;
  int c = col % CC;
  int dy = t / 3 - 1, dx = t % 3 - 1;
  int b = m / LL, hw = m % LL;
  int h = hw / WW + dy, w = hw % WW + dx;
  v8bf val;
  if (h >= 0 && h < HH && w >= 0 && w < WW) {
    val = *(const v8bf*)(src + ((size_t)(b * LL + h * WW + w) * CC + c));
  } else {
#pragma unroll
    for (int e = 0; e < 8; ++e) val[e] = f2bf(0.0f);
  }
  *(v8bf*)(dst + (size_t)m * (9 * CC) + col) = val;
}

// ---------------- WMMA GEMM:  C(M,N) = A(M,K) @ W(N,K)^T ----------------
// 128 threads = 4 waves; workgroup tile 64x64; each wave 32x32 via 2x2 WMMA.
// Double-buffered LDS; next tile staged with GLOBAL_LOAD_ASYNC_TO_LDS_B128
// (ASYNCcnt) while the current tile feeds the WMMAs.
// Epilogues:
//  0: Cf[m*ldc+n] = acc
//  1: Cb[m*ldc+n] = bf16(relu(acc + bias[n]))
//  2: out[b, choff+n, hw]  = relu(acc + bias[n])   (NCHW write)
//  3: out[b, choff+n, hw] += relu(acc + bias[n])   (NCHW add)
//  4: out[b, choff+n, hw] += acc                   (NCHW add, no act)
//  5: Cf[m*ldc+n] = softplus(acc + bias[n])
__global__ __launch_bounds__(128) void gemm_bf16_kernel(
    const bf16_t* __restrict__ A, const bf16_t* __restrict__ W, int M, int N,
    int K, int epi, int ldc, float* __restrict__ Cf, bf16_t* __restrict__ Cb,
    const float* __restrict__ bias, float* __restrict__ Op, int choff) {
  __shared__ __align__(16) bf16_t As[2][64 * 40];
  __shared__ __align__(16) bf16_t Bs[2][64 * 40];
  const int tid = threadIdx.x;
  const int lane = tid & 31;
  const int wv = tid >> 5;
  const int wm = wv >> 1, wn = wv & 1;
  const int m0 = blockIdx.x * 64;
  const int n0 = blockIdx.y * 64;

  v8f acc[2][2];
#pragma unroll
  for (int i = 0; i < 2; ++i)
#pragma unroll
    for (int j = 0; j < 2; ++j)
#pragma unroll
      for (int e = 0; e < 8; ++e) acc[i][j][e] = 0.0f;

  auto load_tile = [&](int k0, int buf) {
    bf16_t* as = &As[buf][0];
    bf16_t* bs = &Bs[buf][0];
#if HAVE_ASYNC_LDS
    if (n0 + 64 <= N) {   // full tile: DMA straight into LDS (ASYNCcnt)
#pragma unroll
      for (int s = tid; s < 256; s += 128) {
        int row = s >> 2;
        int col = (s & 3) << 3;
        __builtin_amdgcn_global_load_async_to_lds_b128(
            (AS1 v4i*)(A + (size_t)(m0 + row) * K + k0 + col),
            (AS3 v4i*)(as + row * 40 + col), 0, 0);
        __builtin_amdgcn_global_load_async_to_lds_b128(
            (AS1 v4i*)(W + (size_t)(n0 + row) * K + k0 + col),
            (AS3 v4i*)(bs + row * 40 + col), 0, 0);
      }
      return;
    }
#endif
#pragma unroll
    for (int s = tid; s < 256; s += 128) {
      int row = s >> 2;
      int col = (s & 3) << 3;
      *(v8bf*)(as + row * 40 + col) =
          *(const v8bf*)(A + (size_t)(m0 + row) * K + k0 + col);
      int nrow = n0 + row;
      v8bf wb;
      if (nrow < N) {
        wb = *(const v8bf*)(W + (size_t)nrow * K + k0 + col);
      } else {
#pragma unroll
        for (int e = 0; e < 8; ++e) wb[e] = f2bf(0.0f);
      }
      *(v8bf*)(bs + row * 40 + col) = wb;
    }
  };

  load_tile(0, 0);
  wait_async_lds();
  __syncthreads();

  int cur = 0;
  for (int k0 = 0; k0 < K; k0 += 32) {
    if (k0 + 32 < K) load_tile(k0 + 32, cur ^ 1);
#pragma unroll
    for (int j = 0; j < 2; ++j) {
      v16bf bfb = load_frag(&Bs[cur][(wn * 32 + j * 16) * 40], 40, lane);
#pragma unroll
      for (int i = 0; i < 2; ++i) {
        v16bf afa = load_frag(&As[cur][(wm * 32 + i * 16) * 40], 40, lane);
        acc[i][j] = __builtin_amdgcn_wmma_f32_16x16x32_bf16(
            false, afa, false, bfb, (short)0, acc[i][j], false, false);
      }
    }
    wait_async_lds();
    __syncthreads();
    cur ^= 1;
  }

  // epilogue; C/D layout: VGPR v, lane -> M = v + (lane>=16 ? 8 : 0), N = lane&15
  int nl = lane & 15;
  int rb = (lane >> 4) * 8;
#pragma unroll
  for (int i = 0; i < 2; ++i)
#pragma unroll
    for (int j = 0; j < 2; ++j) {
      int mbase = m0 + wm * 32 + i * 16 + rb;
      int nn = n0 + wn * 32 + j * 16 + nl;
      if (nn >= N) continue;
#pragma unroll
      for (int v = 0; v < 8; ++v) {
        int m = mbase + v;
        float a = acc[i][j][v];
        if (epi == 0) {
          Cf[(size_t)m * ldc + nn] = a;
        } else if (epi == 1) {
          float t = a + bias[nn];
          t = t > 0.0f ? t : 0.0f;
          Cb[(size_t)m * ldc + nn] = f2bf(t);
        } else if (epi == 2 || epi == 3) {
          float t = a + bias[nn];
          t = t > 0.0f ? t : 0.0f;
          int b = m / LL, hw = m % LL;
          size_t oi = ((size_t)b * (2 * CC) + choff + nn) * LL + hw;
          if (epi == 2) Op[oi] = t; else Op[oi] += t;
        } else if (epi == 4) {
          int b = m / LL, hw = m % LL;
          size_t oi = ((size_t)b * (2 * CC) + choff + nn) * LL + hw;
          Op[oi] += a;
        } else {  // 5: softplus(acc + bias)
          float t = a + bias[nn];
          float sp = (t > 20.0f) ? t : logf(1.0f + __expf(t));
          Cf[(size_t)m * ldc + nn] = sp;
        }
      }
    }
}

// ---------------- depthwise causal conv1d (K=4) + SiLU ----------------
__global__ void conv1d_silu_kernel(const float* __restrict__ xz,
                                   const float* __restrict__ w,
                                   const float* __restrict__ bias,
                                   float* __restrict__ xm32,
                                   bf16_t* __restrict__ xmb) {
  int i = blockIdx.x * blockDim.x + threadIdx.x;
  if (i >= MM * DIN) return;
  int d = i % DIN;
  int bl = i / DIN;
  int l = bl % LL;
  float acc = bias[d];
#pragma unroll
  for (int j = 0; j < 4; ++j) {
    int ls = l - 3 + j;
    if (ls >= 0) acc += w[d * 4 + j] * xz[(size_t)(bl - 3 + j) * (2 * DIN) + d];
  }
  float sv = acc / (1.0f + __expf(-acc));
  xm32[i] = sv;
  xmb[i] = f2bf(sv);
}

// ---------------- dt-rank pad + cast (K 16 -> 32) ----------------
__global__ void dtpad_kernel(const float* __restrict__ xdbl,
                             bf16_t* __restrict__ dtb) {
  int i = blockIdx.x * blockDim.x + threadIdx.x;
  if (i >= MM * 32) return;
  int k = i % 32, m = i / 32;
  dtb[i] = (k < DTR) ? f2bf(xdbl[(size_t)m * 48 + k]) : f2bf(0.0f);
}

// ---------------- selective scan ----------------
// lane = state n (16 lanes per (b,d)); 2048 (b,d) pairs; 16-lane reduction.
__global__ __launch_bounds__(256) void scan_kernel(
    const float* __restrict__ dt, const float* __restrict__ u,
    const float* __restrict__ xdbl, const float* __restrict__ A_log,
    const float* __restrict__ Dp, float* __restrict__ y) {
  int g = blockIdx.x * 256 + threadIdx.x;
  int n = g & 15;
  int pair = g >> 4;           // 0..2047
  int b = pair >> 9;           // / 512
  int d = pair & (DIN - 1);
  float Aa = -__expf(A_log[d * DST + n]);
  float Dv = Dp[d];
  float h = 0.0f;
  size_t base = (size_t)b * LL;
  for (int l = 0; l < LL; ++l) {
    size_t r = base + l;
    if (l + 1 < LL) {          // hide next-step latency (global_prefetch_b8)
      __builtin_prefetch(&dt[(r + 1) * DIN + d], 0, 0);
      __builtin_prefetch(&xdbl[(r + 1) * 48 + DTR + n], 0, 0);
    }
    float dtv = dt[r * DIN + d];
    float uv = u[r * DIN + d];
    float Bv = xdbl[r * 48 + DTR + n];
    float Cv = xdbl[r * 48 + DTR + DST + n];
    h = h * __expf(dtv * Aa) + dtv * Bv * uv;
    float yv = h * Cv;
#pragma unroll
    for (int o = 8; o > 0; o >>= 1) yv += __shfl_xor(yv, o, 16);
    if (n == 0) y[r * DIN + d] = yv + uv * Dv;
  }
}

// ---------------- gate: y *= silu(z); cast to bf16 ----------------
__global__ void gate_kernel(const float* __restrict__ xz,
                            const float* __restrict__ y,
                            bf16_t* __restrict__ yb) {
  int i = blockIdx.x * blockDim.x + threadIdx.x;
  if (i >= MM * DIN) return;
  int d = i % DIN;
  int bl = i / DIN;
  float z = xz[(size_t)bl * (2 * DIN) + DIN + d];
  float gg = z / (1.0f + __expf(-z));
  yb[i] = f2bf(y[i] * gg);
}

// ---------------- workspace layout (bytes) ----------------
#define OFF_XB   ((size_t)0)                      // (M,256) bf16   4,718,592
#define OFF_HS   (OFF_XB + 4718592)               // (M,256) bf16
#define OFF_T2B  (OFF_HS + 4718592)               // (M,256) bf16 conv mid
#define OFF_IMXZ (OFF_T2B + 4718592)              // im2col (M,2304)bf16 / xz (M,1024)f32
#define OFF_XM32 (OFF_IMXZ + 42467328)            // (M,512) f32
#define OFF_XMB  (OFF_XM32 + 18874368)            // (M,512) bf16 (reused as yb)
#define OFF_XDBL (OFF_XMB + 9437184)              // (M,48)  f32
#define OFF_DTB  (OFF_XDBL + 1769472)             // (M,32)  bf16
#define OFF_DT   (OFF_DTB + 589824)               // (M,512) f32
#define OFF_Y    (OFF_DT + 18874368)              // (M,512) f32
#define OFF_W1   (OFF_Y + 18874368)               // 512x256 bf16
#define OFF_W2   (OFF_W1 + 262144)                // 256x2304 bf16
#define OFF_W3   (OFF_W2 + 1179648)
#define OFF_WIN  (OFF_W3 + 1179648)               // 1024x256 bf16
#define OFF_WXP  (OFF_WIN + 524288)               // 48x512 bf16
#define OFF_WDT  (OFF_WXP + 49152)                // 512x32 bf16
#define OFF_WOUT (OFF_WDT + 32768)                // 256x512 bf16
#define WS_NEED  (OFF_WOUT + 262144)              // ~128.5 MB

extern "C" void kernel_launch(void* const* d_in, const int* in_sizes, int n_in,
                              void* d_out, int out_size, void* d_ws,
                              size_t ws_size, hipStream_t stream) {
  (void)in_sizes; (void)n_in; (void)out_size; (void)ws_size;
  const float* x        = (const float*)d_in[0];
  const float* conv1_w  = (const float*)d_in[1];
  const float* conv1_b  = (const float*)d_in[2];
  const float* conv2_w  = (const float*)d_in[3];
  const float* conv2_b  = (const float*)d_in[4];
  const float* conv3_w  = (const float*)d_in[5];
  const float* conv3_b  = (const float*)d_in[6];
  const float* ln_g     = (const float*)d_in[7];
  const float* ln_b     = (const float*)d_in[8];
  const float* in_proj_w = (const float*)d_in[9];
  const float* convd_w  = (const float*)d_in[10];
  const float* convd_b  = (const float*)d_in[11];
  const float* xproj_w  = (const float*)d_in[12];
  const float* dtproj_w = (const float*)d_in[13];
  const float* dtproj_b = (const float*)d_in[14];
  const float* A_log    = (const float*)d_in[15];
  const float* Dp       = (const float*)d_in[16];
  const float* outproj_w = (const float*)d_in[17];
  float* out = (float*)d_out;
  char* ws = (char*)d_ws;

  bf16_t* XB   = (bf16_t*)(ws + OFF_XB);
  bf16_t* HS   = (bf16_t*)(ws + OFF_HS);
  bf16_t* T2B  = (bf16_t*)(ws + OFF_T2B);
  bf16_t* IM   = (bf16_t*)(ws + OFF_IMXZ);
  float*  XZ   = (float*)(ws + OFF_IMXZ);
  float*  XM32 = (float*)(ws + OFF_XM32);
  bf16_t* XMB  = (bf16_t*)(ws + OFF_XMB);
  bf16_t* YB   = (bf16_t*)(ws + OFF_XMB);   // reused after xproj consumes XMB
  float*  XDBL = (float*)(ws + OFF_XDBL);
  bf16_t* DTB  = (bf16_t*)(ws + OFF_DTB);
  float*  DT   = (float*)(ws + OFF_DT);
  float*  Y    = (float*)(ws + OFF_Y);
  bf16_t* W1B  = (bf16_t*)(ws + OFF_W1);
  bf16_t* W2B  = (bf16_t*)(ws + OFF_W2);
  bf16_t* W3B  = (bf16_t*)(ws + OFF_W3);
  bf16_t* WINB = (bf16_t*)(ws + OFF_WIN);
  bf16_t* WXPB = (bf16_t*)(ws + OFF_WXP);
  bf16_t* WDTB = (bf16_t*)(ws + OFF_WDT);
  bf16_t* WOUTB = (bf16_t*)(ws + OFF_WOUT);

  // --- weight precast ---
  cast_nk_kernel<<<(512 * 256 + 255) / 256, 256, 0, stream>>>(conv1_w, W1B, 512 * 256);
  cast_conv3_kernel<<<(256 * 2304 + 255) / 256, 256, 0, stream>>>(conv2_w, W2B);
  cast_conv3_kernel<<<(256 * 2304 + 255) / 256, 256, 0, stream>>>(conv3_w, W3B);
  cast_nk_kernel<<<(1024 * 256 + 255) / 256, 256, 0, stream>>>(in_proj_w, WINB, 1024 * 256);
  cast_nk_kernel<<<(48 * 512 + 255) / 256, 256, 0, stream>>>(xproj_w, WXPB, 48 * 512);
  cast_dtw_kernel<<<(512 * 32 + 255) / 256, 256, 0, stream>>>(dtproj_w, WDTB);
  cast_nk_kernel<<<(256 * 512 + 255) / 256, 256, 0, stream>>>(outproj_w, WOUTB, 256 * 512);

  // --- layernorm + layout ---
  ln_cast_kernel<<<MM / 8, 256, 0, stream>>>(x, ln_g, ln_b, XB, HS);

  // --- y1 = relu(conv1x1(x)) -> writes all 512 out channels ---
  {
    dim3 g(MM / 64, 512 / 64);
    gemm_bf16_kernel<<<g, 128, 0, stream>>>(XB, W1B, MM, 512, 256, /*epi*/2, 0,
                                            nullptr, nullptr, conv1_b, out, 0);
  }

  // --- y2 = relu(conv3x3(relu(conv3x3(x)))) ; added into out[:, :256] ---
  im2col3x3_kernel<<<(MM * 288 + 255) / 256, 256, 0, stream>>>(XB, IM);
  {
    dim3 g(MM / 64, 256 / 64);
    gemm_bf16_kernel<<<g, 128, 0, stream>>>(IM, W2B, MM, 256, 2304, /*epi*/1, 256,
                                            nullptr, T2B, conv2_b, nullptr, 0);
  }
  im2col3x3_kernel<<<(MM * 288 + 255) / 256, 256, 0, stream>>>(T2B, IM);
  {
    dim3 g(MM / 64, 256 / 64);
    gemm_bf16_kernel<<<g, 128, 0, stream>>>(IM, W3B, MM, 256, 2304, /*epi*/3, 0,
                                            nullptr, nullptr, conv3_b, out, 0);
  }

  // --- mamba: in_proj -> xz (M,1024) f32 ---
  {
    dim3 g(MM / 64, 1024 / 64);
    gemm_bf16_kernel<<<g, 128, 0, stream>>>(HS, WINB, MM, 1024, 256, /*epi*/0, 1024,
                                            XZ, nullptr, nullptr, nullptr, 0);
  }

  // --- depthwise causal conv1d + SiLU ---
  conv1d_silu_kernel<<<(MM * DIN + 255) / 256, 256, 0, stream>>>(XZ, convd_w,
                                                                 convd_b, XM32, XMB);

  // --- x_proj -> (M,48) f32 ---
  {
    dim3 g(MM / 64, 1);
    gemm_bf16_kernel<<<g, 128, 0, stream>>>(XMB, WXPB, MM, 48, 512, /*epi*/0, 48,
                                            XDBL, nullptr, nullptr, nullptr, 0);
  }

  // --- dt_proj (K padded to 32) -> softplus -> DT (M,512) ---
  dtpad_kernel<<<(MM * 32 + 255) / 256, 256, 0, stream>>>(XDBL, DTB);
  {
    dim3 g(MM / 64, 512 / 64);
    gemm_bf16_kernel<<<g, 128, 0, stream>>>(DTB, WDTB, MM, 512, 32, /*epi*/5, 512,
                                            DT, nullptr, dtproj_b, nullptr, 0);
  }

  // --- selective scan ---
  scan_kernel<<<128, 256, 0, stream>>>(DT, XM32, XDBL, A_log, Dp, Y);

  // --- gate with silu(z), cast to bf16 ---
  gate_kernel<<<(MM * DIN + 255) / 256, 256, 0, stream>>>(XZ, Y, YB);

  // --- out_proj ; added into out[:, 256:512] ---
  {
    dim3 g(MM / 64, 256 / 64);
    gemm_bf16_kernel<<<g, 128, 0, stream>>>(YB, WOUTB, MM, 256, 512, /*epi*/4, 0,
                                            nullptr, nullptr, nullptr, out, 256);
  }
}
